// KLDivNoTruthLoss_85383949845128
// MI455X (gfx1250) — matrix operand order repaired and checked by reading, not compiled, not run on hardware
//
#include <hip/hip_runtime.h>
#include <stdint.h>

// ---------------------------------------------------------------------------
// KLDivNoTruthLoss for MI455X (gfx1250, wave32, WMMA + async global->LDS)
//   probs = softmax(x/T)+1e-8  (f32 math, stored f16 for matrix cores)
//   t[j]  = mean_c probs[j]*log(probs[j])
//   cross = (P @ P^T)/C   via v_wmma_f32_16x16x32_f16 with double-buffered
//           GLOBAL_LOAD_ASYNC_TO_LDS_B128 staging (ASYNCcnt pipeline),
//   loss  = sum_{i!=j, label eq} (t[j]-cross[i,j])^2 / B   (fused epilogue)
// ---------------------------------------------------------------------------

typedef __attribute__((ext_vector_type(16))) _Float16 v16h;
typedef __attribute__((ext_vector_type(8)))  float    v8f;

union Frag16 { v16h v; uint4 u4[2]; };

#define TILE       128
#define KSTEP      32
#define NTHREADS   256
#define LDS_ROW_U4 5   // 32 halfs = 4 uint4, padded to 5 (80B) to spread banks

__global__ void zero_out_kernel(float* out) { out[0] = 0.0f; }

// ---------------- softmax + self-entropy, f32 -> f16 --------------------
__global__ __launch_bounds__(NTHREADS)
void softmax_prep_kernel(const float* __restrict__ x,
                         _Float16* __restrict__ P,
                         float* __restrict__ t,
                         int C, float invT)
{
    __shared__ float red[8];
    const int row  = blockIdx.x;
    const int tid  = threadIdx.x;
    const int lane = tid & 31;
    const int wave = tid >> 5;
    const float* xr = x + (size_t)row * C;

    // --- row max ---
    float m = -3.4e38f;
    for (int c = tid; c < C; c += NTHREADS) m = fmaxf(m, xr[c]);
    for (int off = 16; off; off >>= 1) m = fmaxf(m, __shfl_xor(m, off, 32));
    if (lane == 0) red[wave] = m;
    __syncthreads();
    if (wave == 0) {
        float mm = red[lane & 7];
        for (int off = 4; off; off >>= 1) mm = fmaxf(mm, __shfl_xor(mm, off, 32));
        if (lane == 0) red[0] = mm;
    }
    __syncthreads();
    m = red[0];
    __syncthreads();

    // --- sum of exp((x-max)/T) ---
    float s = 0.0f;
    for (int c = tid; c < C; c += NTHREADS) s += __expf((xr[c] - m) * invT);
    for (int off = 16; off; off >>= 1) s += __shfl_xor(s, off, 32);
    if (lane == 0) red[wave] = s;
    __syncthreads();
    if (wave == 0) {
        float ss = red[lane & 7];
        for (int off = 4; off; off >>= 1) ss += __shfl_xor(ss, off, 32);
        if (lane == 0) red[0] = ss;
    }
    __syncthreads();
    s = red[0];
    __syncthreads();
    const float inv_s = 1.0f / s;

    // --- probs (store f16) + t accumulation ---
    float tl = 0.0f;
    _Float16* Pr = P + (size_t)row * C;
    for (int c = tid; c < C; c += NTHREADS) {
        float p = __expf((xr[c] - m) * invT) * inv_s + 1e-8f;
        Pr[c] = (_Float16)p;
        tl += p * __logf(p);
    }
    for (int off = 16; off; off >>= 1) tl += __shfl_xor(tl, off, 32);
    if (lane == 0) red[wave] = tl;
    __syncthreads();
    if (wave == 0) {
        float tt = red[lane & 7];
        for (int off = 4; off; off >>= 1) tt += __shfl_xor(tt, off, 32);
        if (lane == 0) t[row] = tt / (float)C;
    }
}

// ---------------- fused Gram-GEMM + masked squared-KL reduction ----------
// Block: 256 threads = 8 waves, computes a 128x128 tile of cross.
// Wave (wm,wn) in a 2x4 grid owns 64x32 => 4x2 WMMA 16x16 accumulators.
// Global->LDS staging uses async DMA (no VGPR landing), double buffered.
__global__ __launch_bounds__(NTHREADS)
void gram_loss_kernel(const _Float16* __restrict__ P,
                      const float*    __restrict__ t,
                      const int*      __restrict__ lab,
                      float*          __restrict__ out,
                      int Brows, int C)
{
    __shared__ uint4 sA[2][TILE * LDS_ROW_U4];   // 2 x 10 KB
    __shared__ uint4 sB[2][TILE * LDS_ROW_U4];   // 2 x 10 KB
    __shared__ int   sLabI[TILE];
    __shared__ int   sLabJ[TILE];
    __shared__ float sT[TILE];
    __shared__ float wred[8];

    const int tid     = threadIdx.x;
    const int lane    = tid & 31;
    const int wave    = tid >> 5;
    const int wm      = wave >> 2;        // 0..1
    const int wn      = wave & 3;         // 0..3
    const int i0      = blockIdx.y * TILE;
    const int j0      = blockIdx.x * TILE;
    const int halfsel = lane >> 4;        // 0 or 1
    const int l16     = lane & 15;

    // this thread's staging assignment: 2 chunks of A + 2 of B per K-step
    const int chunk0 = tid;               // 0..255
    const int chunk1 = tid + NTHREADS;    // 256..511
    const int rowA0  = chunk0 >> 2, cpA0 = chunk0 & 3;
    const int rowA1  = chunk1 >> 2, cpA1 = chunk1 & 3;

    // issue async global->LDS loads for one K-step into buffer `buf`
    auto issue_async = [&](int buf, int k) {
        const char* ga0 = (const char*)(P + (size_t)(i0 + rowA0) * C + k) + cpA0 * 16;
        const char* ga1 = (const char*)(P + (size_t)(i0 + rowA1) * C + k) + cpA1 * 16;
        const char* gb0 = (const char*)(P + (size_t)(j0 + rowA0) * C + k) + cpA0 * 16;
        const char* gb1 = (const char*)(P + (size_t)(j0 + rowA1) * C + k) + cpA1 * 16;
        // low 32 bits of a generic shared-memory address == LDS byte offset
        uint32_t la0 = (uint32_t)(uintptr_t)&sA[buf][rowA0 * LDS_ROW_U4 + cpA0];
        uint32_t la1 = (uint32_t)(uintptr_t)&sA[buf][rowA1 * LDS_ROW_U4 + cpA1];
        uint32_t lb0 = (uint32_t)(uintptr_t)&sB[buf][rowA0 * LDS_ROW_U4 + cpA0];
        uint32_t lb1 = (uint32_t)(uintptr_t)&sB[buf][rowA1 * LDS_ROW_U4 + cpA1];
        asm volatile("global_load_async_to_lds_b128 %0, %1, off"
                     :: "v"(la0), "v"(ga0) : "memory");
        asm volatile("global_load_async_to_lds_b128 %0, %1, off"
                     :: "v"(la1), "v"(ga1) : "memory");
        asm volatile("global_load_async_to_lds_b128 %0, %1, off"
                     :: "v"(lb0), "v"(gb0) : "memory");
        asm volatile("global_load_async_to_lds_b128 %0, %1, off"
                     :: "v"(lb1), "v"(gb1) : "memory");
    };

    // stage labels / t for this tile
    if (tid < TILE) {
        sLabI[tid] = lab[i0 + tid];
        sLabJ[tid] = lab[j0 + tid];
        sT[tid]    = t[j0 + tid];
    }

    v8f acc[4][2];
    #pragma unroll
    for (int m = 0; m < 4; ++m)
        #pragma unroll
        for (int n = 0; n < 2; ++n)
            acc[m][n] = (v8f){};

    const int NIT = C / KSTEP;
    issue_async(0, 0);                       // prologue: fill buffer 0

    for (int it = 0; it < NIT; ++it) {
        const int cur = it & 1;
        // my async writes for `cur` have landed in LDS
        asm volatile("s_wait_asynccnt 0x0" ::: "memory");
        // everyone's writes landed; everyone done reading the other buffer
        // (their previous ds_loads completed before their last WMMA issued)
        __syncthreads();
        // overlap: start DMA for the next K-step into the other buffer
        if (it + 1 < NIT) issue_async(1 - cur, (it + 1) * KSTEP);

        // A fragment (16x32 f16): lane<16 holds K 0-7 & 16-23, lane>=16 K 8-15 & 24-31
        Frag16 a[4], b[2];
        #pragma unroll
        for (int m = 0; m < 4; ++m) {
            const int r = wm * 64 + m * 16 + l16;
            a[m].u4[0] = sA[cur][r * LDS_ROW_U4 + halfsel];       // K halfsel*8 .. +7
            a[m].u4[1] = sA[cur][r * LDS_ROW_U4 + halfsel + 2];   // K 16+halfsel*8 .. +7
        }
        // B fragment (32x16 f16) = rows of P (Gram): same K layout as A
        #pragma unroll
        for (int n = 0; n < 2; ++n) {
            const int r = wn * 32 + n * 16 + l16;
            b[n].u4[0] = sB[cur][r * LDS_ROW_U4 + halfsel];
            b[n].u4[1] = sB[cur][r * LDS_ROW_U4 + halfsel + 2];
        }

        #pragma unroll
        for (int m = 0; m < 4; ++m)
            #pragma unroll
            for (int n = 0; n < 2; ++n)
                acc[m][n] = __builtin_amdgcn_wmma_f32_16x16x32_f16(
                    false, a[m].v, false, b[n].v,
                    (short)0, acc[m][n], false, false);
    }

    // epilogue: C/D layout -> (M = r + 8*halfsel, N = l16) within 16x16 tile
    const float invC = 1.0f / (float)C;
    float local = 0.0f;
    #pragma unroll
    for (int m = 0; m < 4; ++m) {
        const int mBase = wm * 64 + m * 16 + halfsel * 8;  // tile-local row base
        #pragma unroll
        for (int n = 0; n < 2; ++n) {
            const int nLoc = wn * 32 + n * 16 + l16;       // tile-local col
            const int gn   = j0 + nLoc;
            const float tj = sT[nLoc];
            const int   lj = sLabJ[nLoc];
            #pragma unroll
            for (int r = 0; r < 8; ++r) {
                const int mLoc  = mBase + r;
                const int gm    = i0 + mLoc;
                const float cv  = acc[m][n][r] * invC;
                const float kl  = tj - cv;
                const bool  hit = (sLabI[mLoc] == lj) && (gm != gn);
                local += hit ? kl * kl : 0.0f;
            }
        }
    }

    for (int off = 16; off; off >>= 1) local += __shfl_xor(local, off, 32);
    if (lane == 0) wred[wave] = local;
    __syncthreads();
    if (tid == 0) {
        float tot = 0.0f;
        #pragma unroll
        for (int w = 0; w < 8; ++w) tot += wred[w];
        atomicAdd(out, tot / (float)Brows);
    }
}

// ---------------------------------------------------------------------------
extern "C" void kernel_launch(void* const* d_in, const int* in_sizes, int n_in,
                              void* d_out, int out_size, void* d_ws, size_t ws_size,
                              hipStream_t stream)
{
    const float* x   = (const float*)d_in[0];   // [B, C] logits
    const int*   tgt = (const int*)d_in[1];     // [B] labels
    float*       out = (float*)d_out;           // scalar

    const int Btot = in_sizes[1];               // 8192
    const int C    = in_sizes[0] / Btot;        // 1024

    _Float16* P = (_Float16*)d_ws;                                  // 16 MB f16 probs
    float*    t = (float*)((char*)d_ws + (size_t)Btot * C * sizeof(_Float16));

    zero_out_kernel<<<1, 1, 0, stream>>>(out);
    softmax_prep_kernel<<<Btot, NTHREADS, 0, stream>>>(x, P, t, C, 0.25f /* 1/T */);

    dim3 grid(Btot / TILE, Btot / TILE);        // 64 x 64 tiles
    gram_loss_kernel<<<grid, NTHREADS, 0, stream>>>(P, t, tgt, out, Btot, C);
}
